// AdaptiveHighPassFilter_111669149831
// MI455X (gfx1250) — matrix operand order repaired and checked by reading, not compiled, not run on hardware
//
#include <hip/hip_runtime.h>
#include <hip/hip_bf16.h>

typedef __attribute__((ext_vector_type(16))) _Float16 v16h;
typedef __attribute__((ext_vector_type(8)))  _Float16 v8h;
typedef __attribute__((ext_vector_type(8)))  float    v8f;
typedef __attribute__((ext_vector_type(4)))  float    v4f;
typedef __attribute__((ext_vector_type(4)))  int      v4i;

#define B_  8
#define C_  64
#define H_  128
#define W_  128
#define CH_ 32    // C/2 = conv1 output channels

// ---- CDNA5 async global->LDS copy (ASYNCcnt-tracked), with fallback -------
#if defined(__gfx1250__) && \
    __has_builtin(__builtin_amdgcn_global_load_async_to_lds_b128) && \
    __has_builtin(__builtin_amdgcn_s_wait_asynccnt)
#define HAVE_ASYNC_LDS 1
#else
#define HAVE_ASYNC_LDS 0
#endif

__device__ __forceinline__ void copy16_g2lds(const _Float16* __restrict__ g,
                                             _Float16* __restrict__ l) {
#if HAVE_ASYNC_LDS
  // clang-22 prototype: (v4i* gsrc, v4i* ldst, imm offset, imm cpol)
  __builtin_amdgcn_global_load_async_to_lds_b128(
      (v4i*)g, (v4i*)l, /*offset=*/0, /*cpol=*/0);
#else
  *reinterpret_cast<v8h*>(l) = *reinterpret_cast<const v8h*>(g);
#endif
}

__device__ __forceinline__ void wait_async_then_barrier() {
#if HAVE_ASYNC_LDS
  __builtin_amdgcn_s_wait_asynccnt(0);
#endif
  __syncthreads();
}

// -------------------------------------------------------------------------
// Prep A: w1 (32,64,3,3) f32  ->  w1t[(t*32 + o)*64 + c] f16  (cin-fastest)
// -------------------------------------------------------------------------
__global__ __launch_bounds__(256)
void prep_weights_kernel(const float* __restrict__ w1,
                         _Float16* __restrict__ w1t) {
  int idx = blockIdx.x * 256 + threadIdx.x;
  if (idx >= 9 * 32 * 64) return;
  int c  = idx & 63;
  int to = idx >> 6;
  int o  = to & 31;
  int t  = to >> 5;
  w1t[idx] = (_Float16)w1[o * 576 + c * 9 + t];
}

// -------------------------------------------------------------------------
// Prep B: x (B,C,H,W) f32 -> xf16 (B,H,W,C) f16 (channel-fastest pixels)
// -------------------------------------------------------------------------
__global__ __launch_bounds__(256)
void prep_x_kernel(const float* __restrict__ x,
                   _Float16* __restrict__ xf) {
  const long pix = (long)blockIdx.x * 256 + threadIdx.x;
  const int b    = (int)(pix >> 14);
  const int rem  = (int)(pix & 16383);
  const int y    = rem >> 7;
  const int xc   = rem & 127;

  _Float16 buf[64];
  #pragma unroll
  for (int c = 0; c < 64; ++c)
    buf[c] = (_Float16)x[((long)(b * C_ + c) * H_ + y) * W_ + xc];
  v8h* dst = reinterpret_cast<v8h*>(xf + pix * 64);
  #pragma unroll
  for (int i = 0; i < 8; ++i)
    dst[i] = *reinterpret_cast<const v8h*>(&buf[i * 8]);
}

// -------------------------------------------------------------------------
// Pass 1: h = LeakyReLU(conv3x3(x, w1) + b1), implicit GEMM on WMMA.
//   M = pixels (16 per wave), N = 32 output channels, K = 9 taps * 64 cin.
//   Inputs pre-converted to f16; LDS staged with async global->LDS b128.
//   h stored pixel-major: hout[((b*H + y)*W + x)*32 + cout]  (f32)
// -------------------------------------------------------------------------
__global__ __launch_bounds__(128)
void conv1_wmma_kernel(const _Float16* __restrict__ xf,
                       const _Float16* __restrict__ w1t,
                       const float* __restrict__ b1,
                       float* __restrict__ hout) {
  // xs: input slab rows y-1..y+1, cols x0base-1..x0base+64, channel-fastest:
  //     xs[(r*66 + col)*64 + c]                              (25344 B)
  // ws: weights, cin-contiguous: ws[(t*32 + cout)*64 + cin]  (36864 B)
  __shared__ alignas(16) _Float16 xs[3 * 66 * 64];
  __shared__ alignas(16) _Float16 ws[9 * 32 * 64];

  const int half   = blockIdx.x;   // 0/1 -> which 64-pixel half-row
  const int y      = blockIdx.y;
  const int b      = blockIdx.z;
  const int tid    = threadIdx.x;
  const int x0base = half * 64;

  const v8h zero8 = {};

  // ---- stage weights: straight 36864 B copy (2304 x b128) ----
  for (int i = tid; i < 9 * 32 * 64 / 8; i += 128)
    copy16_g2lds(w1t + i * 8, ws + i * 8);

  // ---- stage input slab (async copy valid span, ds-zero padding cells) ----
  for (int r = 0; r < 3; ++r) {
    const int yy = y - 1 + r;
    _Float16* xrow = xs + r * (66 * 64);
    if (yy < 0 || yy >= H_) {
      // whole slab row is padding: zero 66*64 f16 = 528 x b128
      for (int i = tid; i < 66 * 8; i += 128)
        *reinterpret_cast<v8h*>(xrow + i * 8) = zero8;
    } else {
      int clo = 0, chi = 65;                 // slab cols 0..65 <-> xx-1..xx+64
      if (x0base == 0) {                     // left image edge: col 0 invalid
        clo = 1;
        if (tid < 8) *reinterpret_cast<v8h*>(xrow + tid * 8) = zero8;
      }
      if (x0base + 64 >= W_) {               // right image edge: col 65 invalid
        chi = 64;
        if (tid < 8)
          *reinterpret_cast<v8h*>(xrow + 65 * 64 + tid * 8) = zero8;
      }
      const int ncols = chi - clo + 1;       // 65 or 66 contiguous pixel cols
      const _Float16* gsrc =
          xf + ((long)(b * H_ + yy) * W_ + (x0base - 1 + clo)) * 64;
      _Float16* ldst = xrow + clo * 64;
      const int nch = ncols * 8;             // b128 chunks
      for (int i = tid; i < nch; i += 128)
        copy16_g2lds(gsrc + i * 8, ldst + i * 8);
    }
  }
  wait_async_then_barrier();

  const int wv   = tid >> 5;   // wave id 0..3 -> 16-pixel tile
  const int lane = tid & 31;
  const int m    = lane & 15;  // A-matrix row (pixel within tile)
  const int hi   = lane >> 4;  // lane-half
  const int n    = lane & 15;  // B/C/D column (cout within N-tile)

  // accumulators start at the bias (bias depends only on cout)
  const float bias0 = b1[n];
  const float bias1 = b1[16 + n];
  v8f acc0, acc1;
  #pragma unroll
  for (int vv = 0; vv < 8; ++vv) { acc0[vv] = bias0; acc1[vv] = bias1; }

  union Frag { v16h f; v8h h[2]; };

  for (int t = 0; t < 9; ++t) {
    const int dy      = t / 3;
    const int dx      = t - 3 * dy;
    const int colbase = wv * 16 + m + dx;          // xs column for pixel/tap
    const int abase   = (dy * 66 + colbase) * 64;  // + cin
    #pragma unroll
    for (int cbi = 0; cbi < 2; ++cbi) {
      const int cb = cbi * 32;                     // cin block base
      // A fragment 16x32 f16: halves 0..15 -> cin cb+16*hi+0..15
      Frag A;
      const int ab = abase + cb + hi * 16;
      A.h[0] = *reinterpret_cast<const v8h*>(&xs[ab]);
      A.h[1] = *reinterpret_cast<const v8h*>(&xs[ab + 8]);
      // B fragments 32x16 f16: halves 0..15 -> cin cb+16*hi+0..15
      Frag B0, B1;
      const int wb0 = (t * 32 + n) * 64 + cb + hi * 16;
      B0.h[0] = *reinterpret_cast<const v8h*>(&ws[wb0]);
      B0.h[1] = *reinterpret_cast<const v8h*>(&ws[wb0 + 8]);
      const int wb1 = (t * 32 + 16 + n) * 64 + cb + hi * 16;
      B1.h[0] = *reinterpret_cast<const v8h*>(&ws[wb1]);
      B1.h[1] = *reinterpret_cast<const v8h*>(&ws[wb1 + 8]);
      // D = A*B + C, f32 accumulate
      acc0 = __builtin_amdgcn_wmma_f32_16x16x32_f16(
          false, A.f, false, B0.f, (short)0, acc0, false, false);
      acc1 = __builtin_amdgcn_wmma_f32_16x16x32_f16(
          false, A.f, false, B1.f, (short)0, acc1, false, false);
    }
  }

  // LeakyReLU(0.2) + store.  C/D layout: VGPR vv -> pixel m = vv + 8*hi,
  // channel = n (+16 for second N-tile).
  #pragma unroll
  for (int vv = 0; vv < 8; ++vv) {
    const int mm   = vv + 8 * hi;
    const int xpix = x0base + wv * 16 + mm;
    const long ob  = ((long)(b * H_ + y) * W_ + xpix) * CH_;
    float e0 = acc0[vv]; e0 = (e0 >= 0.f) ? e0 : 0.2f * e0;
    float e1 = acc1[vv]; e1 = (e1 >= 0.f) ? e1 : 0.2f * e1;
    hout[ob + n]      = e0;
    hout[ob + 16 + n] = e1;
  }
}

// -------------------------------------------------------------------------
// Pass 2: per pixel -- conv1x1 (9x32) + softmax over 9 taps,
//         w_hp = delta - w_lp, then depthwise spatially-varying 3x3:
//         out = x + sum_t w_hp[t] * x_shift(t)
// -------------------------------------------------------------------------
__global__ __launch_bounds__(256)
void predict_apply_kernel(const float* __restrict__ x,
                          const float* __restrict__ w2,
                          const float* __restrict__ b2,
                          const float* __restrict__ hbuf,
                          float* __restrict__ out) {
  __shared__ float w2s[9 * 32];
  __shared__ float b2s[9];
  const int tid = threadIdx.x;
  if (tid < 9 * 32) w2s[tid] = w2[tid];
  if (tid < 9)      b2s[tid] = b2[tid];
  __syncthreads();

  const long pix = (long)blockIdx.x * 256 + tid;
  const int b    = (int)(pix >> 14);       // / (128*128)
  const int rem  = (int)(pix & 16383);
  const int y    = rem >> 7;
  const int xcol = rem & 127;

  // 32 hidden activations for this pixel (contiguous, vectorized)
  float hv[32];
  const v4f* hp = reinterpret_cast<const v4f*>(hbuf + pix * CH_);
  #pragma unroll
  for (int i = 0; i < 8; ++i) {
    v4f t = hp[i];
    hv[4 * i + 0] = t.x; hv[4 * i + 1] = t.y;
    hv[4 * i + 2] = t.z; hv[4 * i + 3] = t.w;
  }

  // logits + softmax over the 9 taps
  float lg[9];
  float mx = -1e30f;
  #pragma unroll
  for (int t = 0; t < 9; ++t) {
    float s = b2s[t];
    #pragma unroll
    for (int i = 0; i < 32; ++i) s += w2s[t * 32 + i] * hv[i];
    lg[t] = s;
    mx = fmaxf(mx, s);
  }
  float sum = 0.f;
  #pragma unroll
  for (int t = 0; t < 9; ++t) { lg[t] = __expf(lg[t] - mx); sum += lg[t]; }
  const float inv = 1.f / sum;
  float whp[9];
  #pragma unroll
  for (int t = 0; t < 9; ++t)
    whp[t] = ((t == 4) ? 1.f : 0.f) - lg[t] * inv;

  // depthwise spatially-varying conv + residual, all 64 channels
  for (int c = 0; c < C_; ++c) {
    const float* xc = x + (long)(b * C_ + c) * H_ * W_;
    const float center = xc[y * W_ + xcol];
    float acc = center;                      // residual term
    #pragma unroll
    for (int t = 0; t < 9; ++t) {
      const int dy = t / 3 - 1;
      const int dx = t - 3 * (t / 3) - 1;
      float xv;
      if (t == 4) {
        xv = center;
      } else {
        const int yy = y + dy, xx = xcol + dx;
        xv = (yy >= 0 && yy < H_ && xx >= 0 && xx < W_) ? xc[yy * W_ + xx] : 0.f;
      }
      acc += whp[t] * xv;
    }
    out[((long)(b * C_ + c) * H_ + y) * W_ + xcol] = acc;
  }
}

// -------------------------------------------------------------------------
extern "C" void kernel_launch(void* const* d_in, const int* in_sizes, int n_in,
                              void* d_out, int out_size, void* d_ws, size_t ws_size,
                              hipStream_t stream) {
  const float* x  = (const float*)d_in[0];  // (8,64,128,128)
  const float* w1 = (const float*)d_in[1];  // (32,64,3,3)
  const float* b1 = (const float*)d_in[2];  // (32,)
  const float* w2 = (const float*)d_in[3];  // (9,32,1,1)
  const float* b2 = (const float*)d_in[4];  // (9,)
  float* out = (float*)d_out;               // (8,64,128,128)

  // workspace layout (all 256B aligned):
  //   hbuf : (B,H,W,32) f32   = 16777216 B
  //   w1t  : (9,32,64)  f16   =    36864 B
  //   xf16 : (B,H,W,64) f16   = 16777216 B
  char* wsb = (char*)d_ws;
  float*    hbuf = (float*)wsb;
  _Float16* w1t  = (_Float16*)(wsb + 16777216);
  _Float16* xf16 = (_Float16*)(wsb + 16777216 + 36864);

  const int npix = B_ * H_ * W_;            // 131072

  prep_weights_kernel<<<(9 * 32 * 64 + 255) / 256, 256, 0, stream>>>(w1, w1t);
  prep_x_kernel<<<npix / 256, 256, 0, stream>>>(x, xf16);

  dim3 g1(2, H_, B_);                       // half-row per workgroup
  conv1_wmma_kernel<<<g1, 128, 0, stream>>>(xf16, w1t, b1, hbuf);

  predict_apply_kernel<<<npix / 256, 256, 0, stream>>>(x, w2, b2, hbuf, out);
}
